// Encoder_layer_58042188038161
// MI455X (gfx1250) — matrix operand order repaired and checked by reading, not compiled, not run on hardware
//
#include <hip/hip_runtime.h>

// ---------------------------------------------------------------------------
// Encoder layer (self-attn + cross-attn + top-2 MoE) for MI455X / gfx1250.
// bf16 V_WMMA_F32_16X16X32_BF16 everywhere, fp32 accumulation.
// GEMM: K-step 64, double-buffered LDS tiles streamed with
// GLOBAL_LOAD_ASYNC_TO_LDS_B128; next tile issued before s_wait_asynccnt 6 so
// async copy overlaps WMMA. Per-thread global pointers hoisted + incremented
// (no per-iteration 64-bit address math). Workspace requirement: ~190 MB.
// ---------------------------------------------------------------------------

#define D_MODEL 1024
#define N_HEAD  16
#define D_HEAD  64
#define N_EXP   8
#define D_FF    4096
#define B_SZ    4
#define S_DEC   1024
#define S_TXT   512
#define TOK_DEC (B_SZ * S_DEC)   // 4096
#define TOK_TXT (B_SZ * S_TXT)   // 2048

typedef __attribute__((ext_vector_type(16))) __bf16       v16bf;
typedef __attribute__((ext_vector_type(8)))  float        v8f;
typedef __attribute__((ext_vector_type(8)))  unsigned int v8u;

__device__ __forceinline__ unsigned short f2bf(float x) {
  unsigned int u = __builtin_bit_cast(unsigned int, x);
  u += 0x7FFFu + ((u >> 16) & 1u);
  return (unsigned short)(u >> 16);
}

// Async 16B global -> LDS copy (per-lane addresses), tracked by ASYNCcnt.
__device__ __forceinline__ void async_b128(unsigned ldsAddr, const void* gaddr) {
  asm volatile("global_load_async_to_lds_b128 %0, %1, off"
               :: "v"(ldsAddr), "v"(gaddr) : "memory");
}
__device__ __forceinline__ void wait_async0() {
  asm volatile("s_wait_asynccnt 0" ::: "memory");
}
__device__ __forceinline__ void wait_async6() {  // allow 6 in flight (next tile)
  asm volatile("s_wait_asynccnt 6" ::: "memory");
}

// Fragments per CDNA5 ISA 7.12.2 (wave32). l = lane&15, h = lane>>4.
// A (16-bit 16x32): lane l = row; K = 8h+2v,+1 then 16+8h+2v,+1.
// B from [n][k] LDS rows: lane l = col n; K = 8h+2v,+1 then 16+8h+2v,+1.
// *32 variants: row stride 32 bf16; *64 variants: row stride 64 bf16, ks
// selects which 32-wide K slice.
__device__ __forceinline__ v16bf fragA32(const unsigned short* row, int h) {
  const unsigned int* p = (const unsigned int*)row;
  v8u u;
#pragma unroll
  for (int dw = 0; dw < 4; ++dw) {
    u[dw]     = p[4 * h + dw];
    u[4 + dw] = p[8 + 4 * h + dw];
  }
  return __builtin_bit_cast(v16bf, u);
}
__device__ __forceinline__ v16bf fragB32(const unsigned short* row, int h) {
  const unsigned int* p = (const unsigned int*)row;
  v8u u;
#pragma unroll
  for (int dw = 0; dw < 4; ++dw) {
    u[dw]     = p[4 * h + dw];
    u[4 + dw] = p[8 + 4 * h + dw];
  }
  return __builtin_bit_cast(v16bf, u);
}
__device__ __forceinline__ v16bf fragA64(const unsigned short* row, int h, int ks) {
  const unsigned int* p = (const unsigned int*)row + ks * 16;
  v8u u;
#pragma unroll
  for (int dw = 0; dw < 4; ++dw) {
    u[dw]     = p[4 * h + dw];
    u[4 + dw] = p[8 + 4 * h + dw];
  }
  return __builtin_bit_cast(v16bf, u);
}
__device__ __forceinline__ v16bf fragB64(const unsigned short* row, int h, int ks) {
  const unsigned int* p = (const unsigned int*)row + ks * 16;
  v8u u;
#pragma unroll
  for (int dw = 0; dw < 4; ++dw) {
    u[dw]     = p[4 * h + dw];
    u[4 + dw] = p[8 + 4 * h + dw];
  }
  return __builtin_bit_cast(v16bf, u);
}

// ---------------------------------------------------------------------------
// GEMM: C[M,N] (+)= A[M,K] @ Bt[N,K]^T, bf16 inputs, fp32 accumulate.
// Block 256 threads = 8 waves (4 M x 2 N); each wave 32x32 out (2x2 tiles);
// block tile 128x64; K-step 64 (2 WMMA sub-steps, 8 WMMAs/wave/iter);
// double-buffered async-staged LDS. Requires M%128==0, N%64==0, K%64==0.
// ---------------------------------------------------------------------------
#define EPI_F32       0
#define EPI_BF16      1
#define EPI_RELU_BF16 2
#define EPI_SCALE_ACC 3

template <int EPI>
__global__ __launch_bounds__(256) void gemm_bf16_wmma(
    const unsigned short* __restrict__ A,   // [M,K] bf16
    const unsigned short* __restrict__ Bt,  // [N,K] bf16 (pre-transposed)
    float* __restrict__ Cf, unsigned short* __restrict__ Cb,
    int M, int N, int K,
    const float* __restrict__ rowScale, int scaleStride) {
  __shared__ __align__(16) unsigned short sA[2][128 * 64];  // [m][k] x2 buf
  __shared__ __align__(16) unsigned short sB[2][64 * 64];   // [n][k] x2 buf

  const int tid  = threadIdx.x;
  const int lane = tid & 31;
  const int wave = tid >> 5;
  const int wm   = wave >> 1;   // 0..3
  const int wn   = wave & 1;    // 0..1
  const int l    = lane & 15;
  const int h    = lane >> 4;
  const int bm0  = blockIdx.y * 128;
  const int bn0  = blockIdx.x * 64;

  const unsigned sAoff = (unsigned)(size_t)(&sA[0][0]);
  const unsigned sBoff = (unsigned)(size_t)(&sB[0][0]);

  // Hoisted per-thread chunk pointers; advance by 64 bf16 per K-tile.
  // A tile: 128 rows x 128B = 1024 x16B chunks -> 4/thread
  // B tile:  64 rows x 128B =  512 x16B chunks -> 2/thread
  const unsigned short* aPtr[4];
  unsigned aLds[4];
#pragma unroll
  for (int j = 0; j < 4; ++j) {
    int c = tid + j * 256;
    aPtr[j] = A + (size_t)(bm0 + (c >> 3)) * K + (c & 7) * 8;
    aLds[j] = sAoff + c * 16;
  }
  const unsigned short* bPtr[2];
  unsigned bLds[2];
#pragma unroll
  for (int j = 0; j < 2; ++j) {
    int c = tid + j * 256;
    bPtr[j] = Bt + (size_t)(bn0 + (c >> 3)) * K + (c & 7) * 8;
    bLds[j] = sBoff + c * 16;
  }

  auto issueTile = [&](int buf) {
    const unsigned aBase = (unsigned)buf * (128 * 64 * 2);
    const unsigned bBase = (unsigned)buf * (64 * 64 * 2);
#pragma unroll
    for (int j = 0; j < 4; ++j) {
      async_b128(aLds[j] + aBase, aPtr[j]);
      aPtr[j] += 64;
    }
#pragma unroll
    for (int j = 0; j < 2; ++j) {
      async_b128(bLds[j] + bBase, bPtr[j]);
      bPtr[j] += 64;
    }
  };

  v8f zero = {};
  v8f acc[2][2];
#pragma unroll
  for (int i = 0; i < 2; ++i)
#pragma unroll
    for (int j = 0; j < 2; ++j) acc[i][j] = zero;

  issueTile(0);
  const int nk = K >> 6;
  for (int t = 0; t < nk; ++t) {
    const int buf = t & 1;
    if (t + 1 < nk) {
      issueTile(buf ^ 1);  // stream next tile while computing this one
      wait_async6();       // retire current tile's 6 ops (in-order drain)
    } else {
      wait_async0();
    }
    __syncthreads();

    const unsigned short* rowA0 = &sA[buf][(wm * 32 + l) * 64];
    const unsigned short* rowA1 = &sA[buf][(wm * 32 + 16 + l) * 64];
    const unsigned short* rowB0 = &sB[buf][(wn * 32 + l) * 64];
    const unsigned short* rowB1 = &sB[buf][(wn * 32 + 16 + l) * 64];
#pragma unroll
    for (int ks = 0; ks < 2; ++ks) {
      v16bf a0 = fragA64(rowA0, h, ks);
      v16bf a1 = fragA64(rowA1, h, ks);
      v16bf b0 = fragB64(rowB0, h, ks);
      v16bf b1 = fragB64(rowB1, h, ks);
      acc[0][0] = __builtin_amdgcn_wmma_f32_16x16x32_bf16(
          false, a0, false, b0, (short)0, acc[0][0], false, false);
      acc[0][1] = __builtin_amdgcn_wmma_f32_16x16x32_bf16(
          false, a0, false, b1, (short)0, acc[0][1], false, false);
      acc[1][0] = __builtin_amdgcn_wmma_f32_16x16x32_bf16(
          false, a1, false, b0, (short)0, acc[1][0], false, false);
      acc[1][1] = __builtin_amdgcn_wmma_f32_16x16x32_bf16(
          false, a1, false, b1, (short)0, acc[1][1], false, false);
    }
    __syncthreads();
  }

  // C/D layout: lane n = l, VGPR r -> row = r + 8h
#pragma unroll
  for (int i = 0; i < 2; ++i)
#pragma unroll
    for (int j = 0; j < 2; ++j) {
      const int mBase = bm0 + wm * 32 + i * 16 + 8 * h;
      const int nCol  = bn0 + wn * 32 + j * 16 + l;
#pragma unroll
      for (int r = 0; r < 8; ++r) {
        size_t idx = (size_t)(mBase + r) * N + nCol;
        float v = acc[i][j][r];
        if (EPI == EPI_F32) {
          Cf[idx] = v;
        } else if (EPI == EPI_BF16) {
          Cb[idx] = f2bf(v);
        } else if (EPI == EPI_RELU_BF16) {
          Cb[idx] = f2bf(fmaxf(v, 0.0f));
        } else {
          Cf[idx] += rowScale[(size_t)(mBase + r) * scaleStride] * v;
        }
      }
    }
}

// ---------------------------------------------------------------------------
// Flash attention, one wave per (b, head, 16-q tile). bf16 in, bf16 out.
// Q/K tiles async-staged; V transposed via uint -> 2x b16 LDS stores.
// Per 32-key chunk: 4 score WMMAs, LDS online softmax, 4 PV WMMAs.
// ---------------------------------------------------------------------------
__global__ __launch_bounds__(32) void attn_wmma(
    const unsigned short* __restrict__ Q, const unsigned short* __restrict__ K,
    const unsigned short* __restrict__ V, unsigned short* __restrict__ O,
    int Sq, int Sk) {
  __shared__ __align__(16) unsigned short sQ[16 * 64];
  __shared__ __align__(16) unsigned short sK[32 * 64];   // [key][d]
  __shared__ __align__(16) unsigned short sVt[64 * 32];  // [d][key]
  __shared__ float sS[16 * 32];
  __shared__ __align__(16) unsigned short sP[16 * 32];
  __shared__ float rowF[16];
  __shared__ float rowL[16];

  const int lane = threadIdx.x;
  const int l    = lane & 15;
  const int h    = lane >> 4;
  const int b    = blockIdx.z;
  const int hcol = blockIdx.y * D_HEAD;
  const size_t qrow0 = (size_t)b * Sq + (size_t)blockIdx.x * 16;
  const size_t krow0 = (size_t)b * Sk;

  const unsigned sQoff = (unsigned)(size_t)(&sQ[0]);
  const unsigned sKoff = (unsigned)(size_t)(&sK[0]);

  // Q tile: 16 rows x 128B = 128 x16B chunks -> 4 per lane
#pragma unroll
  for (int j = 0; j < 4; ++j) {
    int c = lane + j * 32;
    async_b128(sQoff + c * 16,
               Q + (qrow0 + (c >> 3)) * D_MODEL + hcol + (c & 7) * 8);
  }
  wait_async0();
  __syncthreads();

  v16bf qa[2];
#pragma unroll
  for (int ks = 0; ks < 2; ++ks) qa[ks] = fragA64(sQ + l * 64, h, ks);

  // Hoisted per-lane chunk pointers, advanced by 32 rows per key-chunk.
  const unsigned short* kPtr[8];
#pragma unroll
  for (int j = 0; j < 8; ++j) {
    int c = lane + j * 32;
    kPtr[j] = K + (krow0 + (c >> 3)) * D_MODEL + hcol + (c & 7) * 8;
  }
  const unsigned short* vPtr = V + (krow0 + lane) * D_MODEL + hcol;

  v8f zero = {};
  v8f accO[4];
#pragma unroll
  for (int t = 0; t < 4; ++t) accO[t] = zero;
  float mreg = -3.0e38f, lreg = 0.0f;  // valid on lanes 0..15

  for (int kc = 0; kc < Sk; kc += 32) {
    // K chunk: 32 rows x 128B = 256 chunks -> 8 async per lane
#pragma unroll
    for (int j = 0; j < 8; ++j) {
      async_b128(sKoff + (lane + j * 32) * 16, kPtr[j]);
      kPtr[j] += 32 * D_MODEL;
    }
    // V chunk transposed: lane owns key row `lane`, 32 uints = 64 d-values
    {
      const unsigned int* vu = (const unsigned int*)vPtr;
#pragma unroll
      for (int j = 0; j < 32; ++j) {
        unsigned int w = vu[j];
        sVt[(2 * j) * 32 + lane]     = (unsigned short)w;
        sVt[(2 * j + 1) * 32 + lane] = (unsigned short)(w >> 16);
      }
      vPtr += 32 * D_MODEL;
    }
    wait_async0();
    __syncthreads();

    // S = Q @ K^T  (two 16-key tiles)
#pragma unroll
    for (int t = 0; t < 2; ++t) {
      v8f accS = zero;
#pragma unroll
      for (int ks = 0; ks < 2; ++ks) {
        v16bf bfv = fragB64(sK + (t * 16 + l) * 64, h, ks);
        accS = __builtin_amdgcn_wmma_f32_16x16x32_bf16(
            false, qa[ks], false, bfv, (short)0, accS, false, false);
      }
#pragma unroll
      for (int r = 0; r < 8; ++r)
        sS[(r + 8 * h) * 32 + t * 16 + l] = accS[r] * 0.125f;  // 1/sqrt(64)
    }
    __syncthreads();

    if (lane < 16) {  // online softmax, lane owns row
      float cm = -3.0e38f;
      for (int j = 0; j < 32; ++j) cm = fmaxf(cm, sS[lane * 32 + j]);
      float mn = fmaxf(mreg, cm);
      float f  = __expf(mreg - mn);
      float cs = 0.0f;
      for (int j = 0; j < 32; ++j) {
        float p = __expf(sS[lane * 32 + j] - mn);
        cs += p;
        sP[lane * 32 + j] = f2bf(p);
      }
      lreg = lreg * f + cs;
      mreg = mn;
      rowF[lane] = f;
    }
    __syncthreads();

#pragma unroll
    for (int t = 0; t < 4; ++t)
#pragma unroll
      for (int r = 0; r < 8; ++r) accO[t][r] *= rowF[r + 8 * h];

    v16bf pf = fragA32(sP + l * 32, h);
#pragma unroll
    for (int t = 0; t < 4; ++t) {
      v16bf bfv = fragB32(sVt + (t * 16 + l) * 32, h);
      accO[t] = __builtin_amdgcn_wmma_f32_16x16x32_bf16(
          false, pf, false, bfv, (short)0, accO[t], false, false);
    }
    __syncthreads();
  }

  if (lane < 16) rowL[lane] = lreg;
  __syncthreads();
#pragma unroll
  for (int t = 0; t < 4; ++t)
#pragma unroll
    for (int r = 0; r < 8; ++r)
      O[(qrow0 + r + 8 * h) * D_MODEL + hcol + t * 16 + l] =
          f2bf(accO[t][r] / rowL[r + 8 * h]);
}

// ---------------------------------------------------------------------------
// out = rmsnorm(a + b) * g ; fp32 out (residual stream) + optional bf16 copy.
// ---------------------------------------------------------------------------
__global__ __launch_bounds__(256) void rmsnorm_res(
    const float* __restrict__ a, const float* __restrict__ b,
    const float* __restrict__ g, float* __restrict__ outF,
    unsigned short* __restrict__ outB, int rows) {
  int wave = threadIdx.x >> 5;
  int lane = threadIdx.x & 31;
  int row  = blockIdx.x * 8 + wave;
  if (row >= rows) return;
  const float* pa = a + (size_t)row * D_MODEL;
  const float* pb = b + (size_t)row * D_MODEL;
  float v[32];
  float ss = 0.0f;
#pragma unroll
  for (int j = 0; j < 32; ++j) {
    int c = lane + j * 32;
    v[j] = pa[c] + pb[c];
    ss += v[j] * v[j];
  }
#pragma unroll
  for (int off = 16; off > 0; off >>= 1) ss += __shfl_xor(ss, off, 32);
  float r = rsqrtf(ss * (1.0f / D_MODEL) + 1e-6f);
#pragma unroll
  for (int j = 0; j < 32; ++j) {
    int c = lane + j * 32;
    float o = v[j] * r * g[c];
    outF[(size_t)row * D_MODEL + c] = o;
    if (outB) outB[(size_t)row * D_MODEL + c] = f2bf(o);
  }
}

// ---------------------------------------------------------------------------
// Router: softmax(x @ Wg), top-2 (JAX tie-break), renormalized weights.
// ---------------------------------------------------------------------------
__global__ __launch_bounds__(256) void router(
    const float* __restrict__ x, const float* __restrict__ Wg,
    float* __restrict__ wOut, float* __restrict__ iOut,
    float* __restrict__ combine, int T) {
  int t = blockIdx.x * blockDim.x + threadIdx.x;
  if (t >= T) return;
  float logit[N_EXP];
#pragma unroll
  for (int e = 0; e < N_EXP; ++e) logit[e] = 0.0f;
  const float* px = x + (size_t)t * D_MODEL;
  for (int k = 0; k < D_MODEL; ++k) {
    float xv = px[k];
#pragma unroll
    for (int e = 0; e < N_EXP; ++e) logit[e] += xv * Wg[k * N_EXP + e];
  }
  float mx = logit[0];
#pragma unroll
  for (int e = 1; e < N_EXP; ++e) mx = fmaxf(mx, logit[e]);
  float p[N_EXP], s = 0.0f;
#pragma unroll
  for (int e = 0; e < N_EXP; ++e) { p[e] = __expf(logit[e] - mx); s += p[e]; }
#pragma unroll
  for (int e = 0; e < N_EXP; ++e) p[e] /= s;
  int i1 = 0; float v1 = p[0];
#pragma unroll
  for (int e = 1; e < N_EXP; ++e) if (p[e] > v1) { v1 = p[e]; i1 = e; }
  int i2 = (i1 == 0) ? 1 : 0; float v2 = p[i2];
#pragma unroll
  for (int e = 0; e < N_EXP; ++e)
    if (e != i1 && p[e] > v2) { v2 = p[e]; i2 = e; }
  float sn = v1 + v2;
  float w1 = v1 / sn, w2 = v2 / sn;
  wOut[t * 2] = w1;  wOut[t * 2 + 1] = w2;
  iOut[t * 2] = (float)i1;  iOut[t * 2 + 1] = (float)i2;
#pragma unroll
  for (int e = 0; e < N_EXP; ++e) combine[t * N_EXP + e] = 0.0f;
  combine[t * N_EXP + i1] = w1;
  combine[t * N_EXP + i2] = w2;
}

__global__ void conv_bf16(const float* __restrict__ src,
                          unsigned short* __restrict__ dst, size_t n) {
  size_t i = (size_t)blockIdx.x * blockDim.x + threadIdx.x;
  size_t st = (size_t)gridDim.x * blockDim.x;
  for (; i < n; i += st) dst[i] = f2bf(src[i]);
}

// src [K,N] fp32 -> dst [N,K] bf16  (dst-coalesced)
__global__ void convT_bf16(const float* __restrict__ src,
                           unsigned short* __restrict__ dst, int K, int N) {
  size_t total = (size_t)K * N;
  size_t i = (size_t)blockIdx.x * blockDim.x + threadIdx.x;
  size_t st = (size_t)gridDim.x * blockDim.x;
  for (; i < total; i += st) {
    int n = (int)(i / K), k = (int)(i % K);
    dst[i] = f2bf(src[(size_t)k * N + n]);
  }
}

__global__ void fill_zero(float* __restrict__ p, size_t n) {
  size_t i = (size_t)blockIdx.x * blockDim.x + threadIdx.x;
  size_t st = (size_t)gridDim.x * blockDim.x;
  for (; i < n; i += st) p[i] = 0.0f;
}

// ---------------------------------------------------------------------------
extern "C" void kernel_launch(void* const* d_in, const int* in_sizes, int n_in,
                              void* d_out, int out_size, void* d_ws,
                              size_t ws_size, hipStream_t stream) {
  const float* dec = (const float*)d_in[0];
  const float* txt = (const float*)d_in[1];
  const float* Wsa[8] = {(const float*)d_in[2], (const float*)d_in[3],
                         (const float*)d_in[4], (const float*)d_in[5],
                         (const float*)d_in[6], (const float*)d_in[7],
                         (const float*)d_in[8], (const float*)d_in[9]};
  const float* g  = (const float*)d_in[10];
  const float* Wg = (const float*)d_in[11];
  const float* W1 = (const float*)d_in[12];  // [E, d, dff]
  const float* W2 = (const float*)d_in[13];  // [E, dff, d]
  float* out = (float*)d_out;

  const size_t TD = (size_t)TOK_DEC * D_MODEL;
  const size_t TT = (size_t)TOK_TXT * D_MODEL;
  const size_t DD = (size_t)D_MODEL * D_MODEL;
  const size_t DF = (size_t)D_MODEL * D_FF;

  char* wp = (char*)d_ws;
  auto alloc = [&](size_t bytes) {
    char* r = wp;
    wp += (bytes + 255) & ~(size_t)255;
    return r;
  };
  float* tmp  = (float*)alloc(TD * 4);
  float* x1f  = (float*)alloc(TD * 4);
  float* x2f  = (float*)alloc(TD * 4);
  float* yAcc = (float*)alloc(TD * 4);
  float* comb = (float*)alloc((size_t)TOK_DEC * N_EXP * 4);
  unsigned short* decB = (unsigned short*)alloc(TD * 2);
  unsigned short* txtB = (unsigned short*)alloc(TT * 2);
  unsigned short* WT[8];
  for (int i = 0; i < 8; ++i) WT[i] = (unsigned short*)alloc(DD * 2);
  unsigned short* qB  = (unsigned short*)alloc(TD * 2);
  unsigned short* kB  = (unsigned short*)alloc(TD * 2);
  unsigned short* vB  = (unsigned short*)alloc(TD * 2);
  unsigned short* oB  = (unsigned short*)alloc(TD * 2);
  unsigned short* x1B = (unsigned short*)alloc(TD * 2);
  unsigned short* x2B = (unsigned short*)alloc(TD * 2);
  unsigned short* HB  = (unsigned short*)alloc((size_t)TOK_DEC * D_FF * 2);
  unsigned short* W1T = (unsigned short*)alloc(DF * 2);
  unsigned short* W2T = (unsigned short*)alloc(DF * 2);

  dim3 blk(256);
  auto grid2 = [](int M, int N) { return dim3(N / 64, M / 128); };

  // ---- one-time operand conversion ----
  conv_bf16<<<dim3(2048), blk, 0, stream>>>(dec, decB, TD);
  conv_bf16<<<dim3(1024), blk, 0, stream>>>(txt, txtB, TT);
  for (int i = 0; i < 8; ++i)
    convT_bf16<<<dim3(1024), blk, 0, stream>>>(Wsa[i], WT[i], D_MODEL, D_MODEL);

  // ---- self-attention ----
  gemm_bf16_wmma<EPI_BF16><<<grid2(TOK_DEC, D_MODEL), blk, 0, stream>>>(
      decB, WT[0], nullptr, qB, TOK_DEC, D_MODEL, D_MODEL, nullptr, 0);
  gemm_bf16_wmma<EPI_BF16><<<grid2(TOK_DEC, D_MODEL), blk, 0, stream>>>(
      decB, WT[1], nullptr, kB, TOK_DEC, D_MODEL, D_MODEL, nullptr, 0);
  gemm_bf16_wmma<EPI_BF16><<<grid2(TOK_DEC, D_MODEL), blk, 0, stream>>>(
      decB, WT[2], nullptr, vB, TOK_DEC, D_MODEL, D_MODEL, nullptr, 0);
  attn_wmma<<<dim3(S_DEC / 16, N_HEAD, B_SZ), dim3(32), 0, stream>>>(
      qB, kB, vB, oB, S_DEC, S_DEC);
  gemm_bf16_wmma<EPI_F32><<<grid2(TOK_DEC, D_MODEL), blk, 0, stream>>>(
      oB, WT[3], tmp, nullptr, TOK_DEC, D_MODEL, D_MODEL, nullptr, 0);
  rmsnorm_res<<<dim3(TOK_DEC / 8), blk, 0, stream>>>(tmp, dec, g, x1f, x1B,
                                                     TOK_DEC);

  // ---- cross-attention ----
  gemm_bf16_wmma<EPI_BF16><<<grid2(TOK_DEC, D_MODEL), blk, 0, stream>>>(
      x1B, WT[4], nullptr, qB, TOK_DEC, D_MODEL, D_MODEL, nullptr, 0);
  gemm_bf16_wmma<EPI_BF16><<<grid2(TOK_TXT, D_MODEL), blk, 0, stream>>>(
      txtB, WT[5], nullptr, kB, TOK_TXT, D_MODEL, D_MODEL, nullptr, 0);
  gemm_bf16_wmma<EPI_BF16><<<grid2(TOK_TXT, D_MODEL), blk, 0, stream>>>(
      txtB, WT[6], nullptr, vB, TOK_TXT, D_MODEL, D_MODEL, nullptr, 0);
  attn_wmma<<<dim3(S_DEC / 16, N_HEAD, B_SZ), dim3(32), 0, stream>>>(
      qB, kB, vB, oB, S_DEC, S_TXT);
  gemm_bf16_wmma<EPI_F32><<<grid2(TOK_DEC, D_MODEL), blk, 0, stream>>>(
      oB, WT[7], tmp, nullptr, TOK_DEC, D_MODEL, D_MODEL, nullptr, 0);
  rmsnorm_res<<<dim3(TOK_DEC / 8), blk, 0, stream>>>(tmp, x1f, g, x2f, x2B,
                                                     TOK_DEC);

  // ---- MoE ----
  router<<<dim3(TOK_DEC / 256), blk, 0, stream>>>(
      x2f, Wg, out, out + (size_t)TOK_DEC * 2, comb, TOK_DEC);
  fill_zero<<<dim3(1024), blk, 0, stream>>>(yAcc, TD);
  for (int e = 0; e < N_EXP; ++e) {
    convT_bf16<<<dim3(2048), blk, 0, stream>>>(
        W1 + (size_t)e * DF, W1T, D_MODEL, D_FF);   // [d,dff] -> [dff,d]
    gemm_bf16_wmma<EPI_RELU_BF16><<<grid2(TOK_DEC, D_FF), blk, 0, stream>>>(
        x2B, W1T, nullptr, HB, TOK_DEC, D_FF, D_MODEL, nullptr, 0);
    convT_bf16<<<dim3(2048), blk, 0, stream>>>(
        W2 + (size_t)e * DF, W2T, D_FF, D_MODEL);   // [dff,d] -> [d,dff]
    gemm_bf16_wmma<EPI_SCALE_ACC><<<grid2(TOK_DEC, D_MODEL), blk, 0, stream>>>(
        HB, W2T, yAcc, nullptr, TOK_DEC, D_MODEL, D_FF, comb + e, N_EXP);
  }
  rmsnorm_res<<<dim3(TOK_DEC / 8), blk, 0, stream>>>(
      yAcc, x2f, g, out + (size_t)TOK_DEC * 4, nullptr, TOK_DEC);
}
